// MambaBlock_36129264894627
// MI455X (gfx1250) — matrix operand order repaired and checked by reading, not compiled, not run on hardware
//
#include <hip/hip_runtime.h>
#include <hip/hip_bf16.h>
#include <math.h>

// ---- problem constants (match reference) ----
#define D_MODEL 1024
#define D_STATE 16
#define D_CONV  4
#define D_INNER 2048
#define B_SZ    2
#define SEQ     2048
#define ROWS    (B_SZ * SEQ)        // 4096 token rows
#define E_PROJ  (2 * D_STATE + 1)   // 33

typedef __bf16 bf16;
typedef __attribute__((ext_vector_type(16))) __bf16 v16bf;
typedef __attribute__((ext_vector_type(8)))  float  v8f;

// =====================================================================
// 1) LayerNorm(x) -> bf16 activations for the WMMA in_proj GEMM
// =====================================================================
__global__ void ln_bf16_kernel(const float* __restrict__ x,
                               const float* __restrict__ g,
                               const float* __restrict__ b,
                               bf16* __restrict__ xn) {
  __shared__ float red[256];
  const int row = blockIdx.x;
  const float* xr = x + (size_t)row * D_MODEL;
  float v[4], s = 0.f, s2 = 0.f;
#pragma unroll
  for (int i = 0; i < 4; ++i) {
    v[i] = xr[threadIdx.x + i * 256];
    s += v[i]; s2 += v[i] * v[i];
  }
  red[threadIdx.x] = s; __syncthreads();
  for (int off = 128; off; off >>= 1) {
    if (threadIdx.x < off) red[threadIdx.x] += red[threadIdx.x + off];
    __syncthreads();
  }
  const float mean = red[0] * (1.f / D_MODEL);
  __syncthreads();
  red[threadIdx.x] = s2; __syncthreads();
  for (int off = 128; off; off >>= 1) {
    if (threadIdx.x < off) red[threadIdx.x] += red[threadIdx.x + off];
    __syncthreads();
  }
  const float var  = red[0] * (1.f / D_MODEL) - mean * mean;
  const float rstd = rsqrtf(var + 1e-5f);
#pragma unroll
  for (int i = 0; i < 4; ++i) {
    const int c = threadIdx.x + i * 256;
    xn[(size_t)row * D_MODEL + c] = (bf16)((v[i] - mean) * rstd * g[c] + b[c]);
  }
}

// =====================================================================
// 2) fp32 -> bf16 weight cast
// =====================================================================
__global__ void cvt_bf16_kernel(const float* __restrict__ in,
                                bf16* __restrict__ out, int n) {
  int i = blockIdx.x * 256 + threadIdx.x;
  if (i < n) out[i] = (bf16)in[i];
}

// =====================================================================
// 3) Register-tiled WMMA GEMM:  C[MxN] = A[MxK] * W[NxK]^T (+ R)
//    block = 256 thr (8 waves) -> 128(M) x 256(N) macrotile
//    wave  = 64x64 tile = 4x4 accumulators, 16 v_wmma per 32-K step.
//    A-fragment loads are software-pipelined (rotating 2-deep) so the
//    s_wait for A[i+1] overlaps the 4 WMMAs consuming A[i].
//    A frag: row = lane&15, K in {h*8..h*8+7, 16+h*8..} (h = lane>>4)
//    B frag: col = lane&15, K in {h*8..h*8+7, 16+h*8..} (h = lane>>4)
//    D:      m = (lane>>4)*8 + vgpr, n = lane&15
// =====================================================================
__global__ __launch_bounds__(256)
void gemm_bf16_wmma_kernel(const bf16* __restrict__ A,
                           const bf16* __restrict__ W,
                           float* __restrict__ C,
                           const float* __restrict__ R,  // optional residual
                           int M, int N, int K) {
  const int lane = threadIdx.x & 31;
  const int wave = threadIdx.x >> 5;
  const int wm = wave >> 2, wn = wave & 3;
  const int baseM = blockIdx.y * 128 + wm * 64;
  const int baseN = blockIdx.x * 256 + wn * 64;
  const int half = lane >> 4;
  const int l15  = lane & 15;

  const v8f vzero = {0.f, 0.f, 0.f, 0.f, 0.f, 0.f, 0.f, 0.f};
  v8f acc[4][4];
#pragma unroll
  for (int i = 0; i < 4; ++i)
#pragma unroll
    for (int j = 0; j < 4; ++j) acc[i][j] = vzero;

  union Frag { uint4 q[2]; v16bf v; };

  // per-lane base pointers (row component is loop-invariant)
  const bf16* aRow[4];
#pragma unroll
  for (int i = 0; i < 4; ++i)
    aRow[i] = A + (size_t)(baseM + i * 16 + l15) * K + half * 8;
  const bf16* wRow[4];
#pragma unroll
  for (int j = 0; j < 4; ++j)
    wRow[j] = W + (size_t)(baseN + j * 16 + l15) * K + half * 8;

  for (int k0 = 0; k0 < K; k0 += 32) {
    // B fragments for this K-slab (reused by all 4 A row-tiles)
    Frag bfr[4];
#pragma unroll
    for (int j = 0; j < 4; ++j) {
      const bf16* p = wRow[j] + k0;
      bfr[j].q[0] = *(const uint4*)p;
      bfr[j].q[1] = *(const uint4*)(p + 16);
    }
    if (k0 + 32 < K) {  // WGP-scope prefetch: fill L0/L2 for next K-slab
      __builtin_prefetch(aRow[0] + k0 + 32, 0, 3);
      __builtin_prefetch(wRow[0] + k0 + 32, 0, 3);
    }

    // software-pipelined A fragments
    Frag aCur;
    {
      const bf16* p = aRow[0] + k0;
      aCur.q[0] = *(const uint4*)p;
      aCur.q[1] = *(const uint4*)(p + 16);
    }
#pragma unroll
    for (int i = 0; i < 4; ++i) {
      Frag aNext;
      if (i < 3) {
        const bf16* p = aRow[i + 1] + k0;
        aNext.q[0] = *(const uint4*)p;          // issued before WMMAs on aCur
        aNext.q[1] = *(const uint4*)(p + 16);
      }
#pragma unroll
      for (int j = 0; j < 4; ++j)
        acc[i][j] = __builtin_amdgcn_wmma_f32_16x16x32_bf16(
            false, aCur.v, false, bfr[j].v, (short)0, acc[i][j], false, false);
      if (i < 3) aCur = aNext;
    }
  }

#pragma unroll
  for (int i = 0; i < 4; ++i)
#pragma unroll
    for (int j = 0; j < 4; ++j) {
      const int n = baseN + j * 16 + l15;
#pragma unroll
      for (int vv = 0; vv < 8; ++vv) {
        const int m = baseM + i * 16 + half * 8 + vv;
        const size_t idx = (size_t)m * N + n;
        float val = acc[i][j][vv];
        if (R) val += R[idx];
        C[idx] = val;
      }
    }
}

// =====================================================================
// 4) depthwise causal conv (D_CONV=4) over xz[:, :, 0:D_INNER] + SiLU
// =====================================================================
__global__ void conv_silu_kernel(const float* __restrict__ xz,
                                 const float* __restrict__ cw,
                                 const float* __restrict__ cb,
                                 float* __restrict__ xs) {
  const int idx = blockIdx.x * 256 + threadIdx.x;   // over ROWS*D_INNER
  if (idx >= ROWS * D_INNER) return;
  const int d   = idx & (D_INNER - 1);
  const int row = idx >> 11;
  const int b   = row >> 11;            // row / SEQ
  const int l   = row & (SEQ - 1);
  float acc = cb[d];
#pragma unroll
  for (int k = 0; k < D_CONV; ++k) {
    const int lp = l + k - (D_CONV - 1);
    if (lp >= 0)
      acc += xz[(size_t)(b * SEQ + lp) * (2 * D_INNER) + d] * cw[d * D_CONV + k];
  }
  xs[idx] = acc / (1.f + __expf(-acc));   // SiLU
}

// =====================================================================
// 5) x_proj: x_dbl[row, 0:33] = xs[row,:] . x_proj_w[e,:]
// =====================================================================
__global__ void xproj_kernel(const float* __restrict__ xs,
                             const float* __restrict__ w,
                             float* __restrict__ xd) {
  const int row  = blockIdx.x;
  const int lane = threadIdx.x & 31;
  const int wave = threadIdx.x >> 5;
  const float* xr = xs + (size_t)row * D_INNER;
  for (int e = wave; e < E_PROJ; e += 8) {
    const float* wr = w + (size_t)e * D_INNER;
    float acc = 0.f;
    for (int i = lane; i < D_INNER; i += 32) acc = fmaf(xr[i], wr[i], acc);
    for (int off = 16; off; off >>= 1) acc += __shfl_xor(acc, off, 32);
    if (lane == 0) xd[(size_t)row * E_PROJ + e] = acc;
  }
}

// =====================================================================
// 6) delta pre-pass: delta[row,d] = softplus(x_dbl[row,0]*dtw[d]+dtb[d])
// =====================================================================
__global__ void delta_kernel(const float* __restrict__ xd,
                             const float* __restrict__ dtw,
                             const float* __restrict__ dtb,
                             float* __restrict__ delta) {
  const int idx = blockIdx.x * 256 + threadIdx.x;
  if (idx >= ROWS * D_INNER) return;
  const int d   = idx & (D_INNER - 1);
  const int row = idx >> 11;
  const float t = xd[(size_t)row * E_PROJ] * dtw[d] + dtb[d];
  delta[idx] = (t > 20.f) ? t : log1pf(__expf(t));
}

// =====================================================================
// 7) selective scan: thread per (b, d, n); 16-lane shfl_xor reduction
// =====================================================================
__global__ void scan_kernel(const float* __restrict__ xs,
                            const float* __restrict__ delta,
                            const float* __restrict__ xd,
                            const float* __restrict__ A_log,
                            float* __restrict__ ysc) {
  const int gid = blockIdx.x * 256 + threadIdx.x;   // 0..65535
  const int n  = gid & (D_STATE - 1);
  const int ch = gid >> 4;                          // 0..4095
  const int b  = ch >> 11;                          // / D_INNER
  const int d  = ch & (D_INNER - 1);
  const float A_dn = -__expf(A_log[d * D_STATE + n]);
  float h = 0.f;
  for (int t = 0; t < SEQ; ++t) {
    const size_t row  = (size_t)b * SEQ + t;
    const size_t rd   = row * D_INNER + d;
    const float dt = delta[rd];
    const float xt = xs[rd];
    const float Bn = xd[row * E_PROJ + 1 + n];
    const float Cn = xd[row * E_PROJ + 1 + D_STATE + n];
    const float dA = __expf(dt * A_dn);
    h = fmaf(dA, h, dt * xt * Bn);
    float y = h * Cn;
    y += __shfl_xor(y, 1, 32);
    y += __shfl_xor(y, 2, 32);
    y += __shfl_xor(y, 4, 32);
    y += __shfl_xor(y, 8, 32);
    if (n == 0) ysc[rd] = y;
  }
}

// =====================================================================
// 8) gate: y = (y_scan + xs*D) * silu(z)  -> bf16 for out_proj GEMM
// =====================================================================
__global__ void gate_kernel(const float* __restrict__ ysc,
                            const float* __restrict__ xs,
                            const float* __restrict__ Dp,
                            const float* __restrict__ xz,
                            bf16* __restrict__ ybf) {
  const int idx = blockIdx.x * 256 + threadIdx.x;
  if (idx >= ROWS * D_INNER) return;
  const int d   = idx & (D_INNER - 1);
  const int row = idx >> 11;
  const float y = ysc[idx] + xs[idx] * Dp[d];
  const float z = xz[(size_t)row * (2 * D_INNER) + D_INNER + d];
  const float sz = z / (1.f + __expf(-z));
  ybf[idx] = (bf16)(y * sz);
}

// =====================================================================
// launcher
// =====================================================================
extern "C" void kernel_launch(void* const* d_in, const int* in_sizes, int n_in,
                              void* d_out, int out_size, void* d_ws, size_t ws_size,
                              hipStream_t stream) {
  const float* x       = (const float*)d_in[0];   // (B,SEQ,D_MODEL)
  const float* inpw    = (const float*)d_in[1];   // (4096,1024)
  const float* convw   = (const float*)d_in[2];   // (2048,4)
  const float* convb   = (const float*)d_in[3];   // (2048)
  const float* xprojw  = (const float*)d_in[4];   // (33,2048)
  const float* dtw     = (const float*)d_in[5];   // (2048,1)
  const float* dtb     = (const float*)d_in[6];   // (2048)
  const float* A_log   = (const float*)d_in[7];   // (2048,16)
  const float* Dp      = (const float*)d_in[8];   // (2048)
  const float* outpw   = (const float*)d_in[9];   // (1024,2048)
  const float* normg   = (const float*)d_in[10];
  const float* normb   = (const float*)d_in[11];
  float* out = (float*)d_out;                     // (B,SEQ,D_MODEL) fp32

  // ---- workspace layout (all offsets 256B aligned) ----
  char* ws = (char*)d_ws;
  size_t o = 0;
  bf16*  xn_bf   = (bf16*)(ws + o); o += (size_t)ROWS * D_MODEL * 2;       // 8.0 MB
  bf16*  win_bf  = (bf16*)(ws + o); o += (size_t)2 * D_INNER * D_MODEL * 2;// 8.0 MB
  bf16*  wout_bf = (bf16*)(ws + o); o += (size_t)D_MODEL * D_INNER * 2;    // 4.0 MB
  float* xz      = (float*)(ws + o); o += (size_t)ROWS * 2 * D_INNER * 4;  // 64 MB
  float* xs      = (float*)(ws + o); o += (size_t)ROWS * D_INNER * 4;      // 32 MB
  float* xd      = (float*)(ws + o); o += (size_t)ROWS * E_PROJ * 4;       // 0.5 MB
  float* dlt     = (float*)(ws + o); o += (size_t)ROWS * D_INNER * 4;      // 32 MB
  float* ysc     = (float*)(ws + o); o += (size_t)ROWS * D_INNER * 4;      // 32 MB
  bf16*  ybf     = (bf16*)(ws + o); o += (size_t)ROWS * D_INNER * 2;       // 16 MB

  const int elems = ROWS * D_INNER;               // 8,388,608

  // 1) LayerNorm -> bf16
  ln_bf16_kernel<<<ROWS, 256, 0, stream>>>(x, normg, normb, xn_bf);

  // 2) weight casts
  {
    int n1 = 2 * D_INNER * D_MODEL;               // 4,194,304
    cvt_bf16_kernel<<<(n1 + 255) / 256, 256, 0, stream>>>(inpw, win_bf, n1);
    int n2 = D_MODEL * D_INNER;                   // 2,097,152
    cvt_bf16_kernel<<<(n2 + 255) / 256, 256, 0, stream>>>(outpw, wout_bf, n2);
  }

  // 3) in_proj: xz[4096,4096] = xn[4096,1024] * in_proj_w^T  (WMMA bf16)
  {
    dim3 grid((2 * D_INNER) / 256, ROWS / 128);   // (16, 32)
    gemm_bf16_wmma_kernel<<<grid, 256, 0, stream>>>(
        xn_bf, win_bf, xz, nullptr, ROWS, 2 * D_INNER, D_MODEL);
  }

  // 4) depthwise conv + SiLU
  conv_silu_kernel<<<(elems + 255) / 256, 256, 0, stream>>>(xz, convw, convb, xs);

  // 5) x_proj (skinny, wave-dot)
  xproj_kernel<<<ROWS, 256, 0, stream>>>(xs, xprojw, xd);

  // 6) softplus delta pre-pass
  delta_kernel<<<(elems + 255) / 256, 256, 0, stream>>>(xd, dtw, dtb, dlt);

  // 7) selective scan: 65536 threads = (b,d,n)
  scan_kernel<<<(B_SZ * D_INNER * D_STATE) / 256, 256, 0, stream>>>(
      xs, dlt, xd, A_log, ysc);

  // 8) gate -> bf16
  gate_kernel<<<(elems + 255) / 256, 256, 0, stream>>>(ysc, xs, Dp, xz, ybf);

  // 9) out_proj + residual: out[4096,1024] = y[4096,2048]*out_w^T + x  (WMMA)
  {
    dim3 grid(D_MODEL / 256, ROWS / 128);         // (4, 32)
    gemm_bf16_wmma_kernel<<<grid, 256, 0, stream>>>(
        ybf, wout_bf, out, x, ROWS, D_MODEL, D_INNER);
  }
}
